// TransformerEncoder_34059090658019
// MI455X (gfx1250) — matrix operand order repaired and checked by reading, not compiled, not run on hardware
//
#include <hip/hip_runtime.h>
#include <hip/hip_bf16.h>
#include <cstdint>

#define BB    32
#define SS    511
#define CC    768
#define HH    12
#define NDEPTH 8
#define HIDN  3072
#define NN    512
#define DHD   64
#define ATT_SCALE 0.125f
#define KC    256   // K-chunk staged into LDS by the TDM

typedef __attribute__((ext_vector_type(16))) _Float16 v16h;
typedef __attribute__((ext_vector_type(8)))  float    v8f;
typedef __attribute__((ext_vector_type(4)))  unsigned u32x4;
typedef __attribute__((ext_vector_type(8)))  int      i32x8;
typedef __attribute__((ext_vector_type(4)))  int      i32x4;

static __device__ __forceinline__ v8f wmma_f16(v16h a, v16h b, v8f c) {
  // emits v_wmma_f32_16x16x32_f16
  return __builtin_amdgcn_wmma_f32_16x16x32_f16(false, a, false, b, (short)0, c, false, false);
}

// --- Tensor Data Mover: 2D tile (rows x cols f16) global -> LDS -------------
// D# layout per CDNA5 ISA: group0 = {count/type,
// lds_addr, global_addr}, group1 = {data_size, tensor dims, tile dims, stride}.
static __device__ __forceinline__ void tdm_load_2d_f16(
    const _Float16* gaddr, unsigned lds_off, int rows, int cols, int row_stride) {
  unsigned long long ga = (unsigned long long)(uintptr_t)gaddr;
  u32x4 g0;
  g0[0] = 1u;                                             // count=1, user desc
  g0[1] = lds_off;                                        // lds_addr (bytes)
  g0[2] = (unsigned)(ga & 0xFFFFFFFFu);                   // global_addr[31:0]
  g0[3] = (unsigned)((ga >> 32) & 0x1FFFFFFu) | (2u << 30); // [56:32] | type=2
  i32x8 g1;
  g1[0] = (int)(1u << 16);                                // data_size=1 (2B), no mcast
  g1[1] = (int)(((unsigned)cols & 0xFFFFu) << 16);        // tensor_dim0[15:0] @ 63:48
  g1[2] = (int)((((unsigned)cols >> 16) & 0xFFFFu) |      // tensor_dim0[31:16]
                (((unsigned)rows & 0xFFFFu) << 16));      // tensor_dim1[15:0]
  g1[3] = (int)((((unsigned)rows >> 16) & 0xFFFFu) |      // tensor_dim1[31:16]
                (((unsigned)cols & 0xFFFFu) << 16));      // tile_dim0 @ 127:112
  g1[4] = (int)((unsigned)rows & 0xFFFFu);                // tile_dim1 (tile_dim2=0)
  g1[5] = row_stride;                                     // tensor_dim0_stride[31:0]
  g1[6] = 0;                                              // stride hi / dim1_stride lo
  g1[7] = 0;
  i32x4 z4 = {0, 0, 0, 0};
#if __has_include(<hip/amd_detail/amd_gfx1250_TDM.h>)
  i32x8 z8 = {0, 0, 0, 0, 0, 0, 0, 0};                    // therock clang-23: 6-arg form
  __builtin_amdgcn_tensor_load_to_lds(g0, g1, z4, z4, z8, 0);
#else
  __builtin_amdgcn_tensor_load_to_lds(g0, g1, z4, z4, 0); // ROCm 7.2 clang-22: 5-arg
#endif
}

// --- fragment loaders (wave32; contiguous k = 16*(lane/16) + i in A and B) ---

// A: 16x32 f16, rows = M. A points at tile start; works for global or LDS.
static __device__ __forceinline__ v16h ldA(const _Float16* A, int lda, int k0) {
  const int lane = threadIdx.x & 31;
  const _Float16* p = A + (size_t)(lane & 15) * lda + k0 + ((lane >> 4) << 4);
  v16h a;
#pragma unroll
  for (int i = 0; i < 8; ++i) a[i] = p[i];
#pragma unroll
  for (int i = 0; i < 8; ++i) a[i + 8] = p[i + 8];
  return a;
}

// B: 32x16 f16 where B[k][n] = W[n][k]; W row-major, k contiguous.
static __device__ __forceinline__ v16h ldB_t(const _Float16* W, int ldw, int k0) {
  const int lane = threadIdx.x & 31;
  const _Float16* p = W + (size_t)(lane & 15) * ldw + k0 + ((lane >> 4) << 4);
  v16h b;
#pragma unroll
  for (int i = 0; i < 16; ++i) b[i] = p[i];
  return b;
}

// B: 32x16 f16 where B[k][n] = Bm[k*ldb + n] (row-major K x N storage).
static __device__ __forceinline__ v16h ldB_n(const _Float16* Bm, int ldb, int k0) {
  const int lane = threadIdx.x & 31;
  const _Float16* p = Bm + (size_t)(k0 + ((lane >> 4) << 4)) * ldb + (lane & 15);
  v16h b;
#pragma unroll
  for (int i = 0; i < 16; ++i) b[i] = p[(size_t)i * ldb];
  return b;
}

// ----------------------------- kernels ------------------------------------

__global__ void k_embed(const float* __restrict__ x, const float* __restrict__ cls,
                        const float* __restrict__ pos, float* __restrict__ h, long total) {
  long idx = (long)blockIdx.x * blockDim.x + threadIdx.x;
  if (idx >= total) return;
  int c = (int)(idx % CC);
  long t = idx / CC;
  int n = (int)(t % NN);
  int b = (int)(t / NN);
  float v = (n == 0) ? cls[c] : x[((long)b * SS + (n - 1)) * CC + c];
  h[idx] = v + pos[(long)n * CC + c];
}

__global__ void k_maskf(const float* __restrict__ mask, float* __restrict__ mf) {
  int i = blockIdx.x * blockDim.x + threadIdx.x;
  if (i >= BB * NN) return;
  int n = i % NN, b = i / NN;
  mf[i] = (n == 0) ? 1.0f : mask[b * SS + n - 1];
}

__global__ void k_cvt(const float* __restrict__ s, _Float16* __restrict__ d, long n) {
  long i = (long)blockIdx.x * blockDim.x + threadIdx.x;
  if (i < n) d[i] = (_Float16)s[i];
}

__global__ void __launch_bounds__(256)
k_ln(const float* __restrict__ x, const float* __restrict__ g, const float* __restrict__ bt,
     _Float16* __restrict__ y16, float* __restrict__ y32) {
  __shared__ float a1[256], a2[256];
  const int row = blockIdx.x, tid = threadIdx.x;
  const float* xr = x + (size_t)row * CC;
  float s = 0.f, s2 = 0.f;
  for (int c = tid; c < CC; c += 256) { float v = xr[c]; s += v; s2 += v * v; }
  a1[tid] = s; a2[tid] = s2;
  __syncthreads();
  for (int o = 128; o > 0; o >>= 1) {
    if (tid < o) { a1[tid] += a1[tid + o]; a2[tid] += a2[tid + o]; }
    __syncthreads();
  }
  float mu  = a1[0] * (1.0f / CC);
  float var = a2[0] * (1.0f / CC) - mu * mu;
  float rs  = rsqrtf(var + 1e-6f);
  for (int c = tid; c < CC; c += 256) {
    float v = (xr[c] - mu) * rs * g[c] + bt[c];
    if (y16) y16[(size_t)row * CC + c] = (_Float16)v;
    if (y32) y32[(size_t)row * CC + c] = v;
  }
}

// out[m,n] = sum_k A[m,k] * W[n,k] + bias[n]
// One wave computes a 32x64 output block (2 row-tiles x 4 col-tiles, 8 WMMA
// accumulators). A-tile is staged to LDS in 32xKC chunks by the TDM; W is
// streamed from global with contiguous 128-bit loads.
// mode 0: out16 = val   | mode 1: out32 += val | mode 2: out16 = gelu(val)
__global__ void __launch_bounds__(32)
k_gemm(const _Float16* __restrict__ A, const _Float16* __restrict__ W,
       const float* __restrict__ bias, float* out32, _Float16* out16,
       int Nout, int K, int mode) {
  __shared__ _Float16 sA[32 * KC];   // 16 KB per (single-wave) block
  const int n0 = blockIdx.x * 64;
  const int m0 = blockIdx.y * 32;
  const unsigned lds_off = (unsigned)(uintptr_t)&sA[0];

  v8f acc[2][4] = {};
  for (int kc = 0; kc < K; kc += KC) {
    tdm_load_2d_f16(A + (size_t)m0 * K + kc, lds_off, 32, KC, K);
    __builtin_amdgcn_s_wait_tensorcnt(0);
    for (int k0 = 0; k0 < KC; k0 += 32) {
      v16h a0 = ldA(sA, KC, k0);
      v16h a1 = ldA(sA + 16 * KC, KC, k0);
#pragma unroll
      for (int j = 0; j < 4; ++j) {
        v16h bf = ldB_t(W + (size_t)(n0 + 16 * j) * K + kc, K, k0);
        acc[0][j] = wmma_f16(a0, bf, acc[0][j]);
        acc[1][j] = wmma_f16(a1, bf, acc[1][j]);
      }
    }
  }

  const int lane = threadIdx.x & 31;
  const int col = lane & 15;
  const int rb  = (lane >> 4) << 3;
#pragma unroll
  for (int i = 0; i < 2; ++i) {
#pragma unroll
    for (int j = 0; j < 4; ++j) {
      const int n = n0 + 16 * j + col;
      const float bv = bias[n];
#pragma unroll
      for (int r = 0; r < 8; ++r) {
        size_t idx = (size_t)(m0 + 16 * i + rb + r) * Nout + n;
        float v = acc[i][j][r] + bv;
        if (mode == 0)      out16[idx] = (_Float16)v;
        else if (mode == 1) out32[idx] += v;
        else                out16[idx] = (_Float16)(0.5f * v * (1.0f + erff(v * 0.70710678118f)));
      }
    }
  }
}

// attention: one block per (b, head, 16-query tile); 4 waves.
__global__ void __launch_bounds__(128)
k_attn(const _Float16* __restrict__ qkv, const float* __restrict__ maskf,
       _Float16* __restrict__ o16) {
  __shared__ float    sS[16][NN];
  __shared__ _Float16 sP[16][NN];
  __shared__ float    sRed[16][8];

  const int b = blockIdx.z, hd = blockIdx.y, n0 = blockIdx.x * 16;
  const int tid = threadIdx.x, wv = tid >> 5;
  const int lane = tid & 31, col = lane & 15, rb = (lane >> 4) << 3;
  const int ldq = 3 * CC;
  const _Float16* qb = qkv + (size_t)b * NN * ldq + hd * DHD;
  const _Float16* kb = qb + CC;
  const _Float16* vb = qb + 2 * CC;

  // 1) scores S = Q K^T : Q fragments are loop-invariant -> hoist them.
  const _Float16* qtile = qb + (size_t)n0 * ldq;
  const v16h qa0 = ldA(qtile, ldq, 0);
  const v16h qa1 = ldA(qtile, ldq, 32);
  for (int t = 0; t < 8; ++t) {
    int m0 = wv * 128 + t * 16;
    v8f acc = {};
    acc = wmma_f16(qa0, ldB_t(kb + (size_t)m0 * ldq, ldq, 0), acc);
    acc = wmma_f16(qa1, ldB_t(kb + (size_t)m0 * ldq, ldq, 32), acc);
#pragma unroll
    for (int r = 0; r < 8; ++r) sS[rb + r][m0 + col] = acc[r];
  }
  __syncthreads();

  // 2) softmax over keys; mask applied AFTER softmax (matches reference)
  {
    const int r = tid >> 3, c0 = tid & 7;
    float mx = -3.4e38f;
    for (int c = c0; c < NN; c += 8) mx = fmaxf(mx, sS[r][c] * ATT_SCALE);
    sRed[r][c0] = mx;
    __syncthreads();
    if (c0 == 0) {
      float v = sRed[r][0];
      for (int i = 1; i < 8; ++i) v = fmaxf(v, sRed[r][i]);
      sRed[r][0] = v;
    }
    __syncthreads();
    mx = sRed[r][0];
    __syncthreads();
    float sum = 0.f;
    for (int c = c0; c < NN; c += 8) {
      float e = expf(sS[r][c] * ATT_SCALE - mx);
      sS[r][c] = e;
      sum += e;
    }
    sRed[r][c0] = sum;
    __syncthreads();
    if (c0 == 0) {
      float v = 0.f;
      for (int i = 0; i < 8; ++i) v += sRed[r][i];
      sRed[r][0] = v;
    }
    __syncthreads();
    const float inv = 1.0f / sRed[r][0];
    const float* mrow = maskf + (size_t)b * NN;
    for (int c = c0; c < NN; c += 8)
      sP[r][c] = (_Float16)(sS[r][c] * inv * mrow[c]);
  }
  __syncthreads();

  // 3) O = P V : each wave owns a 16-wide slice of DH=64
  const int d0 = wv * 16;
  v8f acc = {};
  for (int m0 = 0; m0 < NN; m0 += 32)
    acc = wmma_f16(ldA(&sP[0][0], NN, m0), ldB_n(vb + d0, ldq, m0), acc);
#pragma unroll
  for (int r = 0; r < 8; ++r)
    o16[((size_t)b * NN + n0 + rb + r) * CC + hd * DHD + d0 + col] = (_Float16)acc[r];
}

// ----------------------------- host side ----------------------------------

extern "C" void kernel_launch(void* const* d_in, const int* in_sizes, int n_in,
                              void* d_out, int out_size, void* d_ws, size_t ws_size,
                              hipStream_t stream) {
  const float* x      = (const float*)d_in[0];
  const float* mask   = (const float*)d_in[1];
  const float* cls    = (const float*)d_in[2];
  const float* pos    = (const float*)d_in[3];
  const float* ln1_g  = (const float*)d_in[4];
  const float* ln1_b  = (const float*)d_in[5];
  const float* qkv_w  = (const float*)d_in[6];
  const float* qkv_b  = (const float*)d_in[7];
  const float* proj_w = (const float*)d_in[8];
  const float* proj_b = (const float*)d_in[9];
  const float* ln2_g  = (const float*)d_in[10];
  const float* ln2_b  = (const float*)d_in[11];
  const float* fc1_w  = (const float*)d_in[12];
  const float* fc1_b  = (const float*)d_in[13];
  const float* fc2_w  = (const float*)d_in[14];
  const float* fc2_b  = (const float*)d_in[15];
  const float* norm_g = (const float*)d_in[16];
  const float* norm_b = (const float*)d_in[17];
  (void)in_sizes; (void)n_in; (void)out_size; (void)ws_size;

  char* ws = (char*)d_ws;
  size_t off = 0;
  auto take = [&](size_t bytes) -> void* {
    void* p = ws + off;
    off += (bytes + 255) & ~(size_t)255;
    return p;
  };
  const size_t M = (size_t)BB * NN;  // 16384 token rows
  float*    h     = (float*)   take(M * CC * 4);
  float*    mf    = (float*)   take((size_t)BB * NN * 4);
  _Float16* y16   = (_Float16*)take(M * CC * 2);
  _Float16* qkv16 = (_Float16*)take(M * 3 * CC * 2);
  _Float16* o16   = (_Float16*)take(M * CC * 2);
  _Float16* hid16 = (_Float16*)take(M * HIDN * 2);
  _Float16* wq16  = (_Float16*)take((size_t)3 * CC * CC * 2);
  _Float16* wp16  = (_Float16*)take((size_t)CC * CC * 2);
  _Float16* w116  = (_Float16*)take((size_t)HIDN * CC * 2);
  _Float16* w216  = (_Float16*)take((size_t)CC * HIDN * 2);

  const long tot = (long)M * CC;
  k_embed<<<(unsigned)((tot + 255) / 256), 256, 0, stream>>>(x, cls, pos, h, tot);
  k_maskf<<<(BB * NN + 255) / 256, 256, 0, stream>>>(mask, mf);

  for (int l = 0; l < NDEPTH; ++l) {
    const long nq = 3L * CC * CC, np = (long)CC * CC, n1 = (long)HIDN * CC;
    k_cvt<<<(unsigned)((nq + 255) / 256), 256, 0, stream>>>(qkv_w + (size_t)l * nq, wq16, nq);
    k_cvt<<<(unsigned)((np + 255) / 256), 256, 0, stream>>>(proj_w + (size_t)l * np, wp16, np);
    k_cvt<<<(unsigned)((n1 + 255) / 256), 256, 0, stream>>>(fc1_w + (size_t)l * n1, w116, n1);
    k_cvt<<<(unsigned)((n1 + 255) / 256), 256, 0, stream>>>(fc2_w + (size_t)l * n1, w216, n1);

    k_ln<<<(unsigned)M, 256, 0, stream>>>(h, ln1_g + (size_t)l * CC, ln1_b + (size_t)l * CC,
                                          y16, nullptr);
    k_gemm<<<dim3(3 * CC / 64, (unsigned)(M / 32)), 32, 0, stream>>>(
        y16, wq16, qkv_b + (size_t)l * 3 * CC, nullptr, qkv16, 3 * CC, CC, 0);
    k_attn<<<dim3(NN / 16, HH, BB), 128, 0, stream>>>(qkv16, mf, o16);
    k_gemm<<<dim3(CC / 64, (unsigned)(M / 32)), 32, 0, stream>>>(
        o16, wp16, proj_b + (size_t)l * CC, h, nullptr, CC, CC, 1);
    k_ln<<<(unsigned)M, 256, 0, stream>>>(h, ln2_g + (size_t)l * CC, ln2_b + (size_t)l * CC,
                                          y16, nullptr);
    k_gemm<<<dim3(HIDN / 64, (unsigned)(M / 32)), 32, 0, stream>>>(
        y16, w116, fc1_b + (size_t)l * HIDN, nullptr, hid16, HIDN, CC, 2);
    k_gemm<<<dim3(CC / 64, (unsigned)(M / 32)), 32, 0, stream>>>(
        hid16, w216, fc2_b + (size_t)l * CC, h, nullptr, CC, HIDN, 1);
  }
  k_ln<<<(unsigned)M, 256, 0, stream>>>(h, norm_g, norm_b, nullptr, (float*)d_out);
}
